// Transformer_3195455668817
// MI455X (gfx1250) — matrix operand order repaired and checked by reading, not compiled, not run on hardware
//
#include <hip/hip_runtime.h>

typedef unsigned short u16;
typedef unsigned int   u32;

typedef __attribute__((ext_vector_type(16))) __bf16 v16bf;
typedef __attribute__((ext_vector_type(8)))  float  v8f;
typedef __attribute__((ext_vector_type(4)))  unsigned int v4u;
typedef __attribute__((ext_vector_type(8)))  int    v8i;
typedef __attribute__((ext_vector_type(4)))  int    v4i;

#define B_   4
#define S_   1024
#define D_   1024
#define H_   8
#define DK_  128
#define DV_  128
#define DFF_ 2048
#define NTOK (B_*S_)
#define MASKV (-1e-30f)

#if defined(__gfx1250__) && __has_builtin(__builtin_amdgcn_tensor_load_to_lds)
#define HAVE_TDM 1
#else
#define HAVE_TDM 0
#endif

// Decisive probe: on the device pass this must hold, otherwise the TDM path is
// silently dead and we want the compiler to tell us.
#if defined(__gfx1250__)
static_assert(HAVE_TDM == 1, "gfx1250 toolchain without __builtin_amdgcn_tensor_load_to_lds");
#endif

__device__ __forceinline__ u16 f2bf(float f){
  u32 u = __float_as_uint(f);
  u32 r = u + 0x7FFFu + ((u >> 16) & 1u);   // round-to-nearest-even
  return (u16)(r >> 16);
}

union AF { v16bf v; u32 w[8]; uint4 q[2]; u16 h[16]; };

#if HAVE_TDM
// LDS byte offset of a __shared__ object: generic LDS address low 32 bits == LDS_ADDR.
__device__ __forceinline__ u32 lds_off(const void* p){
  return (u32)(reinterpret_cast<uintptr_t>(p));
}

// Issue one TDM 2-D tile load: tile (tile0 x tile1) of 2-byte elements from a
// row-major tensor with row stride `stride0` (elements) into LDS at lds_byte,
// padding pad_amount DWORDs after every 2^(padi+1) DWORDs (== one tile row).
__device__ __forceinline__ void tdm_load_tile_2d(
    u32 lds_byte, const u16* gaddr,
    u32 tile0, u32 tile1, u32 stride0, u32 padi, u32 pada)
{
  unsigned long long ga = (unsigned long long)(size_t)gaddr;
  v4u g0;
  g0[0] = 1u;                                              // count=1, user desc
  g0[1] = lds_byte;                                        // lds_addr
  g0[2] = (u32)(ga & 0xFFFFFFFFu);                         // global_addr lo
  g0[3] = (u32)((ga >> 32) & 0x01FFFFFFu) | (2u << 30);    // addr[56:32] | type=2
  v8i g1;
  g1[0] = (int)((1u << 16) | (1u << 20) | ((padi & 7u) << 22) | ((pada & 127u) << 25));
  g1[1] = (int)((tile0 & 0xFFFFu) << 16);                  // tensor_dim0 = tile0
  g1[2] = (int)(((tile0 >> 16) & 0xFFFFu) | ((tile1 & 0xFFFFu) << 16)); // tensor_dim1 = tile1
  g1[3] = (int)(((tile1 >> 16) & 0xFFFFu) | ((tile0 & 0xFFFFu) << 16)); // tile_dim0
  g1[4] = (int)(tile1 & 0xFFFFu);                          // tile_dim1 (tile_dim2=0)
  g1[5] = (int)stride0;                                    // tensor_dim0_stride lo32
  g1[6] = 0;
  g1[7] = 0;
  v4i z4 = {0,0,0,0};
#if defined(__clang_major__) && (__clang_major__ >= 23)
  v8i z8 = {0,0,0,0,0,0,0,0};
  __builtin_amdgcn_tensor_load_to_lds(g0, g1, z4, z4, z8, 0);
#else
  __builtin_amdgcn_tensor_load_to_lds(g0, g1, z4, z4, 0);
#endif
}
#endif  // HAVE_TDM

// ---------------- conversion / layout kernels ----------------
__global__ void k_cvt_bf16(const float* __restrict__ src, u16* __restrict__ dst, int n){
  int i = blockIdx.x * blockDim.x + threadIdx.x;
  if (i < n) dst[i] = f2bf(src[i]);
}

// src: (H, D, DK) -> dst: (D, H*DK) row-major, bf16
__global__ void k_permute_qkv(const float* __restrict__ src, u16* __restrict__ dst){
  int i = blockIdx.x * blockDim.x + threadIdx.x;
  if (i >= D_ * H_ * DK_) return;
  int c = i & (H_*DK_ - 1);
  int d = i >> 10;                 // H_*DK_ == 1024
  int h = c >> 7, k = c & (DK_ - 1);
  dst[i] = f2bf(src[((size_t)h * D_ + d) * DK_ + k]);
}

// ---------------- generic bf16 WMMA GEMM ----------------
// C(MxN) = A(MxK) * B(KxN), bf16 operands, f32 accumulate.
// Block tile 128x128, BK=32. 8 waves in 4(M) x 2(N); each wave 32x64 = 2x4 WMMA tiles.
// Double-buffered: TDM (wave 0) fills buffer p^1 while all waves compute on p;
// one s_wait_tensorcnt + one barrier per K-step.
template<bool BIAS, bool RELU, bool OUTBF>
__global__ __launch_bounds__(256) void k_gemm_bf16(
    const u16* __restrict__ A, const u16* __restrict__ Bm,
    const float* __restrict__ bias,
    float* __restrict__ Cf, u16* __restrict__ Cb,
    int M, int N, int K)
{
  constexpr int BM = 128, BN = 128, BK = 32;
  constexpr int LDA = BK + 8;    // 40 u16 = 16+4 DWORDs  (TDM: padi=3, pada=3)
  constexpr int LDB = BN + 8;    // 136 u16 = 64+4 DWORDs (TDM: padi=5, pada=3)
  __shared__ u16 As[2][BM][LDA];
  __shared__ u16 Bs[2][BK][LDB];

  const int t    = threadIdx.x;
  const int lane = t & 31, wave = t >> 5;
  const int wm   = wave >> 1, wn = wave & 1;
  const int rowBase = blockIdx.y * BM;
  const int colBase = blockIdx.x * BN;

  const v8f vzero = {0.f,0.f,0.f,0.f,0.f,0.f,0.f,0.f};
  v8f acc[2][4];
  #pragma unroll
  for (int mt = 0; mt < 2; ++mt)
    #pragma unroll
    for (int nt = 0; nt < 4; ++nt) acc[mt][nt] = vzero;

  const int mrow0 = wm * 32 + (lane & 15);
  const int khalf = (lane >> 4) * 8;

#if HAVE_TDM
  auto tdm_stage = [&](int k0, int p){
    tdm_load_tile_2d(lds_off(&As[p][0][0]), &A [(size_t)rowBase * K + k0],      BK, BM, (u32)K, 3, 3);
    tdm_load_tile_2d(lds_off(&Bs[p][0][0]), &Bm[(size_t)k0      * N + colBase], BN, BK, (u32)N, 5, 3);
  };
  if (wave == 0) tdm_stage(0, 0);
#endif

  for (int k0 = 0, p = 0; k0 < K; k0 += BK, p ^= 1){
#if HAVE_TDM
    if (wave == 0) __builtin_amdgcn_s_wait_tensorcnt(0);
    __syncthreads();                       // buffer p ready; reads of p^1 done
    if (wave == 0 && k0 + BK < K) tdm_stage(k0 + BK, p ^ 1);
#else
    __syncthreads();
    {
      const int ar = t >> 2, ac = (t & 3) * 8;
      *(uint4*)&As[p][ar][ac]      = *(const uint4*)&A[(size_t)(rowBase + ar)      * K + k0 + ac];
      *(uint4*)&As[p][ar + 64][ac] = *(const uint4*)&A[(size_t)(rowBase + ar + 64) * K + k0 + ac];
      #pragma unroll
      for (int it = 0; it < 2; ++it){
        const int c = t + it * 256;
        const int br = c >> 4, bc = (c & 15) * 8;
        *(uint4*)&Bs[p][br][bc] = *(const uint4*)&Bm[(size_t)(k0 + br) * N + colBase + bc];
      }
    }
    __syncthreads();
#endif

    AF af[2], bfr[4];
    #pragma unroll
    for (int mt = 0; mt < 2; ++mt){
      const int m = mrow0 + mt * 16;
      #pragma unroll
      for (int v = 0; v < 8; ++v){
        const int kk = ((v & 4) ? 16 : 0) + (v & 3) * 2 + khalf;  // ISA 16-bit A layout
        af[mt].w[v] = *(const u32*)&As[p][m][kk];
      }
    }
    #pragma unroll
    for (int nt = 0; nt < 4; ++nt){
      const int n0 = wn * 64 + nt * 16;
      #pragma unroll
      for (int s = 0; s < 16; ++s)
        bfr[nt].h[s] = Bs[p][((s & 8) ? 16 : 0) + khalf + (s & 7)][n0 + (lane & 15)];
    }
    #pragma unroll
    for (int mt = 0; mt < 2; ++mt)
      #pragma unroll
      for (int nt = 0; nt < 4; ++nt)
        acc[mt][nt] = __builtin_amdgcn_wmma_f32_16x16x32_bf16(
            false, af[mt].v, false, bfr[nt].v, (short)0, acc[mt][nt], false, false);
  }

  const int rplus = (lane >> 4) * 8;
  #pragma unroll
  for (int mt = 0; mt < 2; ++mt){
    #pragma unroll
    for (int nt = 0; nt < 4; ++nt){
      const int col = colBase + wn * 64 + nt * 16 + (lane & 15);
      const float bv = BIAS ? bias[col] : 0.f;
      #pragma unroll
      for (int j = 0; j < 8; ++j){
        const int r = rowBase + wm * 32 + mt * 16 + j + rplus;
        float v = acc[mt][nt][j] + bv;
        if (RELU) v = fmaxf(v, 0.f);
        if (OUTBF) Cb[(size_t)r * N + col] = f2bf(v);
        else       Cf[(size_t)r * N + col] = v;
      }
    }
  }
}

// ---------------- flash attention ----------------
// Q,K,V: bf16 (NTOK, H*128) row-major; per (b,h) the head slice starts at col h*128.
// grid = (S/128, B*H), block = 256. Each wave: one 16-row query tile, online softmax.
// V tile staged by TDM; K tile staged manually (needs the d-major transpose).
// Next block's K/V lines prefetched (global_prefetch_b8) during staging.
__global__ __launch_bounds__(256) void k_flash_attn(
    const u16* __restrict__ Q, const u16* __restrict__ Kt, const u16* __restrict__ V,
    const float* __restrict__ mask, u16* __restrict__ O)
{
  __shared__ u16 Ks[128][32];    // [d][key]  (K transposed for QK^T B-operand)
  __shared__ u16 Vs[32][136];    // [key][d], 4-DWORD pad per 64-DWORD row (TDM padi=5,pada=3)
  __shared__ u16 Ps[8][16*32];   // per-wave P scratch (C layout -> A fragment)

  const int t    = threadIdx.x;
  const int lane = t & 31, wave = t >> 5;
  const int bh = blockIdx.y;
  const int b = bh / H_, h = bh % H_;
  const size_t tokBase = (size_t)b * S_;
  const int q0   = blockIdx.x * 128 + wave * 16;
  const int mrow = lane & 15;
  const int khalf = (lane >> 4) * 8;

  // Q tile (16x128) as 4 A-fragments, gathered straight from global
  AF qf[4];
  {
    const u16* qp = Q + (tokBase + q0 + mrow) * (size_t)(H_*DK_) + h * DK_;
    #pragma unroll
    for (int dc = 0; dc < 4; ++dc)
      #pragma unroll
      for (int v = 0; v < 8; ++v){
        const int kk = ((v & 4) ? 16 : 0) + (v & 3) * 2 + khalf;
        qf[dc].w[v] = *(const u32*)(qp + dc * 32 + kk);
      }
  }

  AF ones;                       // all-ones bf16 B fragment for WMMA row sums
  #pragma unroll
  for (int v = 0; v < 8; ++v) ones.w[v] = 0x3F803F80u;

  const v8f vzero = {0.f,0.f,0.f,0.f,0.f,0.f,0.f,0.f};
  v8f Ot[8];
  #pragma unroll
  for (int vt = 0; vt < 8; ++vt) Ot[vt] = vzero;
  float mr[8], lr[8];
  #pragma unroll
  for (int j = 0; j < 8; ++j){ mr[j] = -1e30f; lr[j] = 0.f; }

  for (int j0 = 0; j0 < S_; j0 += 32){
    __syncthreads();             // previous block's LDS reads complete
#if HAVE_TDM
    if (wave == 0)
      tdm_load_tile_2d(lds_off(&Vs[0][0]),
                       V + (tokBase + j0) * (size_t)(H_*DV_) + h * DV_,
                       DV_, 32, (u32)(H_*DV_), 5, 3);
#endif
    // K staged with d-major transpose (and V manually when no TDM)
    #pragma unroll
    for (int it = 0; it < 2; ++it){
      const int c   = t + it * 256;      // 0..511
      const int tok = c >> 4;            // 0..31
      const int d0  = (c & 15) * 8;
      const size_t g = (tokBase + j0 + tok) * (size_t)(H_*DK_) + h * DK_ + d0;
      union { uint4 q; u16 e[8]; } uk; uk.q = *(const uint4*)&Kt[g];
      #pragma unroll
      for (int e = 0; e < 8; ++e) Ks[d0 + e][tok] = uk.e[e];
#if !HAVE_TDM
      *(uint4*)&Vs[tok][d0] = *(const uint4*)&V[g];
#endif
      if (it == 0 && j0 + 32 < S_){      // prefetch next key block (K and V lines)
        __builtin_prefetch((const void*)&Kt[g + (size_t)32 * (H_*DK_)], 0, 0);
        __builtin_prefetch((const void*)&V [g + (size_t)32 * (H_*DV_)], 0, 0);
      }
    }
#if HAVE_TDM
    if (wave == 0) __builtin_amdgcn_s_wait_tensorcnt(0);
#endif
    __syncthreads();

    // S = Q * K^T : two 16x16 tiles, K-dim = 128 in 4 WMMA steps each
    v8f s0 = vzero, s1 = vzero;
    #pragma unroll
    for (int dc = 0; dc < 4; ++dc){
      AF kf0, kf1;
      #pragma unroll
      for (int s = 0; s < 16; ++s){
        const int kr = dc*32 + ((s & 8) ? 16 : 0) + khalf + (s & 7);
        kf0.h[s] = Ks[kr][lane & 15];
        kf1.h[s] = Ks[kr][16 + (lane & 15)];
      }
      s0 = __builtin_amdgcn_wmma_f32_16x16x32_bf16(false, qf[dc].v, false, kf0.v, (short)0, s0, false, false);
      s1 = __builtin_amdgcn_wmma_f32_16x16x32_bf16(false, qf[dc].v, false, kf1.v, (short)0, s1, false, false);
    }

    // mask (per key column)
    const float mv0 = mask[tokBase + j0 + (lane & 15)];
    const float mv1 = mask[tokBase + j0 + 16 + (lane & 15)];
    #pragma unroll
    for (int j = 0; j < 8; ++j){
      s0[j] = mv0 * s0[j] + (1.f - mv0) * MASKV;
      s1[j] = mv1 * s1[j] + (1.f - mv1) * MASKV;
    }

    // online softmax: cross-lane max, then exp + rescale (row = j + 8*(lane>=16))
    #pragma unroll
    for (int j = 0; j < 8; ++j){
      float rm = fmaxf(s0[j], s1[j]);
      #pragma unroll
      for (int d = 1; d < 16; d <<= 1) rm = fmaxf(rm, __shfl_xor(rm, d, 32));
      const float nm = fmaxf(mr[j], rm);
      const float sc = __expf(mr[j] - nm);
      s0[j] = __expf(s0[j] - nm);
      s1[j] = __expf(s1[j] - nm);
      lr[j] *= sc;
      mr[j] = nm;
      #pragma unroll
      for (int vt = 0; vt < 8; ++vt) Ot[vt][j] *= sc;
    }

    // P: C-layout -> A-fragment via per-wave LDS
    u16* ps = &Ps[wave][0];
    {
      const int rpl = (lane >> 4) * 8;
      #pragma unroll
      for (int j = 0; j < 8; ++j){
        const int row = j + rpl;
        ps[row*32 + (lane & 15)]      = f2bf(s0[j]);
        ps[row*32 + 16 + (lane & 15)] = f2bf(s1[j]);
      }
    }
    AF pf;
    #pragma unroll
    for (int v = 0; v < 8; ++v){
      const int kk = ((v & 4) ? 16 : 0) + (v & 3) * 2 + khalf;
      pf.w[v] = *(const u32*)&ps[mrow*32 + kk];
    }

    // row sums of P via WMMA against the all-ones fragment (replaces shuffle ladder)
    v8f sums = __builtin_amdgcn_wmma_f32_16x16x32_bf16(
        false, pf.v, false, ones.v, (short)0, vzero, false, false);
    #pragma unroll
    for (int j = 0; j < 8; ++j) lr[j] += sums[j];

    // O += P * V  (8 column tiles of DV=128)
    #pragma unroll
    for (int vt = 0; vt < 8; ++vt){
      AF vf;
      #pragma unroll
      for (int s = 0; s < 16; ++s)
        vf.h[s] = Vs[((s & 8) ? 16 : 0) + khalf + (s & 7)][vt*16 + (lane & 15)];
      Ot[vt] = __builtin_amdgcn_wmma_f32_16x16x32_bf16(false, pf.v, false, vf.v, (short)0, Ot[vt], false, false);
    }
  }

  // normalize and store concat output (b, s, h*DV + v) as bf16
  const int rpl = (lane >> 4) * 8;
  #pragma unroll
  for (int vt = 0; vt < 8; ++vt){
    const int col = h * DV_ + vt * 16 + (lane & 15);
    #pragma unroll
    for (int j = 0; j < 8; ++j){
      const size_t row = tokBase + q0 + j + rpl;
      O[row * (size_t)(H_*DV_) + col] = f2bf(Ot[vt][j] / lr[j]);
    }
  }
}

// ---------------- fused residual + layernorm ----------------
__global__ __launch_bounds__(256) void k_residual_ln(
    const float* __restrict__ xin, const float* __restrict__ delta,
    const float* __restrict__ gamma, const float* __restrict__ beta, int gb,
    float* __restrict__ xout, u16* __restrict__ xoutb)
{
  __shared__ float red[256];
  const int row = blockIdx.x;
  const int t = threadIdx.x;
  float v[4];
  float s = 0.f, sq = 0.f;
  #pragma unroll
  for (int i = 0; i < 4; ++i){
    const int c = t + i * 256;
    const float x = xin[(size_t)row * D_ + c] + delta[(size_t)row * D_ + c];
    v[i] = x; s += x; sq += x * x;
  }
  red[t] = s; __syncthreads();
  for (int o = 128; o > 0; o >>= 1){ if (t < o) red[t] += red[t + o]; __syncthreads(); }
  const float mean = red[0] * (1.f / D_);
  __syncthreads();
  red[t] = sq; __syncthreads();
  for (int o = 128; o > 0; o >>= 1){ if (t < o) red[t] += red[t + o]; __syncthreads(); }
  const float var = red[0] * (1.f / D_) - mean * mean;
  const float inv = rsqrtf(var + 1e-14f);
  const float g = gamma[gb], be = beta[gb];
  #pragma unroll
  for (int i = 0; i < 4; ++i){
    const int c = t + i * 256;
    const float o = (v[i] - mean) * inv * g + be;
    xout[(size_t)row * D_ + c]  = o;
    xoutb[(size_t)row * D_ + c] = f2bf(o);
  }
}

// ---------------- host orchestration ----------------
extern "C" void kernel_launch(void* const* d_in, const int* in_sizes, int n_in,
                              void* d_out, int out_size, void* d_ws, size_t ws_size,
                              hipStream_t stream)
{
  (void)in_sizes; (void)n_in; (void)out_size; (void)ws_size;

  const float* x     = (const float*)d_in[0];
  const float* mask  = (const float*)d_in[1];
  const float* Wq    = (const float*)d_in[2];
  const float* Wk    = (const float*)d_in[3];
  const float* Wv    = (const float*)d_in[4];
  const float* Wo    = (const float*)d_in[5];
  const float* W1    = (const float*)d_in[6];
  const float* b1    = (const float*)d_in[7];
  const float* W2    = (const float*)d_in[8];
  const float* b2    = (const float*)d_in[9];
  const float* gamma = (const float*)d_in[10];
  const float* beta  = (const float*)d_in[11];

  char* w = (char*)d_ws;
  auto take = [&](size_t bytes) -> char* {
    char* p = w; w += (bytes + 255) & ~(size_t)255; return p;
  };
  u16*   xb   = (u16*)  take((size_t)NTOK * D_ * 2);
  u16*   qb   = (u16*)  take((size_t)NTOK * D_ * 2);
  u16*   kb   = (u16*)  take((size_t)NTOK * D_ * 2);
  u16*   vb   = (u16*)  take((size_t)NTOK * D_ * 2);
  u16*   ob   = (u16*)  take((size_t)NTOK * D_ * 2);
  u16*   h1b  = (u16*)  take((size_t)NTOK * DFF_ * 2);
  u16*   wqb  = (u16*)  take((size_t)D_ * D_ * 2);
  u16*   wkb  = (u16*)  take((size_t)D_ * D_ * 2);
  u16*   wvb  = (u16*)  take((size_t)D_ * D_ * 2);
  u16*   wob  = (u16*)  take((size_t)D_ * D_ * 2);
  u16*   w1b  = (u16*)  take((size_t)D_ * DFF_ * 2);
  u16*   w2b  = (u16*)  take((size_t)DFF_ * D_ * 2);
  float* proj = (float*)take((size_t)NTOK * D_ * 4);
  float* ffn  = (float*)take((size_t)NTOK * D_ * 4);
  float* xcur = (float*)take((size_t)NTOK * D_ * 4);

  const int thr = 256;
  const dim3 gQKVw((D_*H_*DK_ + thr - 1) / thr);
  const dim3 gP(D_ / 128, NTOK / 128);      // N=1024 GEMMs: (8, 32)
  const dim3 g1(DFF_ / 128, NTOK / 128);    // N=2048 GEMM:  (16, 32)
  const dim3 gF(S_ / 128, B_ * H_);

  for (int i = 0; i < 2; ++i){
    k_permute_qkv<<<gQKVw, thr, 0, stream>>>(Wq + (size_t)i*H_*D_*DK_, wqb);
    k_permute_qkv<<<gQKVw, thr, 0, stream>>>(Wk + (size_t)i*H_*D_*DK_, wkb);
    k_permute_qkv<<<gQKVw, thr, 0, stream>>>(Wv + (size_t)i*H_*D_*DV_, wvb);
    k_cvt_bf16<<<(D_*D_ + thr-1)/thr,   thr, 0, stream>>>(Wo + (size_t)i*H_*DV_*D_, wob, D_*D_);
    k_cvt_bf16<<<(D_*DFF_ + thr-1)/thr, thr, 0, stream>>>(W1 + (size_t)i*D_*DFF_,  w1b, D_*DFF_);
    k_cvt_bf16<<<(DFF_*D_ + thr-1)/thr, thr, 0, stream>>>(W2 + (size_t)i*DFF_*D_,  w2b, DFF_*D_);
    if (i == 0)
      k_cvt_bf16<<<(NTOK*D_ + thr-1)/thr, thr, 0, stream>>>(x, xb, NTOK*D_);

    // QKV projections (bf16 out feeds attention directly)
    k_gemm_bf16<false,false,true><<<gP, thr, 0, stream>>>(xb, wqb, nullptr, nullptr, qb, NTOK, D_, D_);
    k_gemm_bf16<false,false,true><<<gP, thr, 0, stream>>>(xb, wkb, nullptr, nullptr, kb, NTOK, D_, D_);
    k_gemm_bf16<false,false,true><<<gP, thr, 0, stream>>>(xb, wvb, nullptr, nullptr, vb, NTOK, D_, D_);

    k_flash_attn<<<gF, thr, 0, stream>>>(qb, kb, vb, mask, ob);

    // output projection + LN1
    k_gemm_bf16<false,false,false><<<gP, thr, 0, stream>>>(ob, wob, nullptr, proj, nullptr, NTOK, D_, D_);
    const float* resid = (i == 0) ? x : xcur;
    k_residual_ln<<<NTOK, thr, 0, stream>>>(resid, proj, gamma, beta, 2*i, xcur, xb);

    // FFN + LN2
    k_gemm_bf16<true,true,true ><<<g1, thr, 0, stream>>>(xb,  w1b, b1 + (size_t)i*DFF_, nullptr, h1b, NTOK, DFF_, D_);
    k_gemm_bf16<true,false,false><<<gP, thr, 0, stream>>>(h1b, w2b, b2 + (size_t)i*D_,  ffn, nullptr, NTOK, D_, DFF_);
    float* lnout = (i == 1) ? (float*)d_out : xcur;
    k_residual_ln<<<NTOK, thr, 0, stream>>>(xcur, ffn, gamma, beta, 2*i + 1, lnout, xb);
  }
}
